// Wave_generatorV2_21973052686539
// MI455X (gfx1250) — matrix operand order repaired and checked by reading, not compiled, not run on hardware
//
#include <hip/hip_runtime.h>
#include <hip/hip_bf16.h>
#include <math.h>
#include <stdint.h>

typedef __attribute__((ext_vector_type(16))) _Float16 v16h;
typedef __attribute__((ext_vector_type(8)))  float    v8f;
typedef int i32x4 __attribute__((vector_size(16)));  // matches builtin's V4i

#define HID 128
#define NHEADS 4
#define N_COE 50
#define N_SCALES 4
#define LAYERS 4
#define SFIX 4096

// ---------------------------------------------------------------------------
// CDNA5 feature shims
// ---------------------------------------------------------------------------
#if defined(__has_builtin)
#if __has_builtin(__builtin_amdgcn_global_load_async_to_lds_b128)
#define HAVE_ASYNC_LDS 1
#endif
#endif
#ifndef HAVE_ASYNC_LDS
#define HAVE_ASYNC_LDS 0
#endif

__device__ __forceinline__ void async_cp16(const _Float16* g, _Float16* l) {
#if HAVE_ASYNC_LDS
  // 16B per lane, global -> LDS through the async path (ASYNCcnt tracked)
  __builtin_amdgcn_global_load_async_to_lds_b128(
      (__attribute__((address_space(1))) i32x4*)(g),
      (__attribute__((address_space(3))) i32x4*)(l), 0, 0);
#else
#pragma unroll
  for (int j = 0; j < 8; ++j) l[j] = g[j];
#endif
}

__device__ __forceinline__ void wait_async0() {
#if HAVE_ASYNC_LDS
#if __has_builtin(__builtin_amdgcn_s_wait_asynccnt)
  __builtin_amdgcn_s_wait_asynccnt(0);
#else
  asm volatile("s_wait_asynccnt 0x0" ::: "memory");
#endif
#endif
}

__device__ __forceinline__ v8f wmma_f16(v16h a, v16h b, v8f c) {
  // D = A(16x32 f16) * B(32x16 f16) + C(16x16 f32)
  return __builtin_amdgcn_wmma_f32_16x16x32_f16(false, a, false, b, (short)0, c,
                                                false, false);
}

// DPP16 butterfly reduction over each 16-lane half (stays in the VALU pipe;
// avoids ds_bpermute traffic in the flash-attention inner loop).
template <int CTRL>
__device__ __forceinline__ float dpp_mov(float x) {
  return __builtin_bit_cast(
      float, __builtin_amdgcn_update_dpp(0, __builtin_bit_cast(int, x), CTRL,
                                         0xf, 0xf, true));
}
__device__ __forceinline__ float dpp_max16(float v) {
  v = fmaxf(v, dpp_mov<0xB1>(v));   // quad_perm(1,0,3,2)  : xor 1
  v = fmaxf(v, dpp_mov<0x4E>(v));   // quad_perm(2,3,0,1)  : xor 2
  v = fmaxf(v, dpp_mov<0x141>(v));  // row_half_mirror     : xor 7 (covers ^4)
  v = fmaxf(v, dpp_mov<0x140>(v));  // row_mirror          : xor 15 (covers ^8)
  return v;
}
__device__ __forceinline__ float dpp_sum16(float v) {
  v += dpp_mov<0xB1>(v);
  v += dpp_mov<0x4E>(v);
  v += dpp_mov<0x141>(v);
  v += dpp_mov<0x140>(v);
  return v;
}

// ---------------------------------------------------------------------------
// 0) Weight f32 -> f16 conversion (once per launch; keeps GEMM staging as
//    pure byte copies so the async-to-LDS path applies to both tiles)
// ---------------------------------------------------------------------------
__global__ void f32_to_f16_kernel(const float* __restrict__ src,
                                  _Float16* __restrict__ dst, int n) {
  int i = blockIdx.x * blockDim.x + threadIdx.x;
  if (i < n) dst[i] = (_Float16)src[i];
}

// ---------------------------------------------------------------------------
// 1) Sine encoding + first GEMM: eig[s, t] = [e, sin(pe), cos(pe)] . W[t, :]
// ---------------------------------------------------------------------------
__global__ void sine_encode_kernel(const float* __restrict__ e,
                                   const float* __restrict__ W,
                                   const float* __restrict__ b,
                                   float* __restrict__ eig) {
  __shared__ float ee[HID + 1];
  const int s = blockIdx.x;
  const int t = threadIdx.x;
  const float ev = e[s];
  if (t == 0) ee[0] = ev;
  if (t < 64) {
    // div_term[j] = exp(2j * (-ln(10000)/128)) = exp(j * -0.14391156...)
    float pe = ev * 100.f * expf((float)t * -0.14391156831f);
    ee[1 + t]  = sinf(pe);
    ee[65 + t] = cosf(pe);
  }
  __syncthreads();
  const float* wr = W + (size_t)t * (HID + 1);
  float acc = b[t];
#pragma unroll 4
  for (int k = 0; k < HID + 1; ++k) acc = fmaf(ee[k], wr[k], acc);
  eig[(size_t)s * HID + t] = acc;
}

// ---------------------------------------------------------------------------
// 2) LayerNorm (128 wide) -> f16 output for WMMA A operands. One wave per row.
// ---------------------------------------------------------------------------
__global__ void layernorm_kernel(const float* __restrict__ x,
                                 const float* __restrict__ g,
                                 const float* __restrict__ bt,
                                 _Float16* __restrict__ out) {
  const int wave = threadIdx.x >> 5;
  const int lane = threadIdx.x & 31;
  const int row  = blockIdx.x * 8 + wave;
  const float* xr = x + (size_t)row * HID;
  float v[4];
  float sum = 0.f;
#pragma unroll
  for (int i = 0; i < 4; ++i) { v[i] = xr[lane + i * 32]; sum += v[i]; }
#pragma unroll
  for (int o = 1; o < 32; o <<= 1) sum += __shfl_xor(sum, o, 32);
  const float mu = sum * (1.f / HID);
  float vs = 0.f;
#pragma unroll
  for (int i = 0; i < 4; ++i) { float d = v[i] - mu; vs += d * d; }
#pragma unroll
  for (int o = 1; o < 32; o <<= 1) vs += __shfl_xor(vs, o, 32);
  const float rstd = rsqrtf(vs * (1.f / HID) + 1e-5f);
#pragma unroll
  for (int i = 0; i < 4; ++i) {
    int c = lane + i * 32;
    out[(size_t)row * HID + c] = (_Float16)((v[i] - mu) * rstd * g[c] + bt[c]);
  }
}

// ---------------------------------------------------------------------------
// 3) WMMA GEMM: out[M,N] = act(A[M,128] @ Wh[N,128]^T + bias (+resid))
//    256 thr = 8 waves; tile 64(M) x 32(N); whole K=128 panel staged once in
//    LDS via async global->LDS b128, then 4 back-to-back WMMAs per wave.
// ---------------------------------------------------------------------------
template <int ACT, int RESID, int OUTF16>
__global__ void gemm_xWT_kernel(const _Float16* __restrict__ A,
                                const _Float16* __restrict__ Wh,
                                const float* __restrict__ bias,
                                const float* __restrict__ resid,
                                void* __restrict__ outp, int N) {
  __shared__ _Float16 lA[64][HID];  // 16 KB
  __shared__ _Float16 lB[32][HID];  //  8 KB
  const int tid  = threadIdx.x;
  const int lane = tid & 31;
  const int w    = tid >> 5;
  const int wm   = w & 3;
  const int wn   = w >> 2;
  const int mbase = blockIdx.y * 64;
  const int nbase = blockIdx.x * 32;
  { // A panel: 64 rows x 128 cols, 4x16B per thread
    const int r  = tid >> 2;
    const int cb = (tid & 3) * 32;
    const _Float16* ap = A + (size_t)(mbase + r) * HID + cb;
#pragma unroll
    for (int j = 0; j < 4; ++j) async_cp16(ap + j * 8, &lA[r][cb + j * 8]);
  }
  { // W panel: 32 rows x 128 cols, 2x16B per thread
    const int r  = tid >> 3;
    const int cb = (tid & 7) * 16;
    const _Float16* wp = Wh + (size_t)(nbase + r) * HID + cb;
#pragma unroll
    for (int j = 0; j < 2; ++j) async_cp16(wp + j * 8, &lB[r][cb + j * 8]);
  }
  wait_async0();
  __syncthreads();

  v8f acc = {};
#pragma unroll
  for (int kc = 0; kc < 4; ++kc) {
    const int k0 = kc * 32;
    v16h af, bf;
    { // A fragment: lanes 0-15 rows m, K in {kb..kb+15}
      const int m  = wm * 16 + (lane & 15);
      const int kb = k0 + ((lane < 16) ? 0 : 16);
#pragma unroll
      for (int i = 0; i < 16; ++i) af[i] = lA[m][kb + i];
    }
    { // B fragment: lane holds col n; lanes 0-15 K=0..15, lanes 16-31 K=16..31
      const int n   = wn * 16 + (lane & 15);
      const int kb2 = k0 + ((lane < 16) ? 0 : 16);
#pragma unroll
      for (int i = 0; i < 16; ++i) bf[i] = lB[n][kb2 + i];
    }
    acc = wmma_f16(af, bf, acc);
  }
  // Epilogue: C layout — lane = col n, VGPR r = row (+8 for upper half-wave)
  const int n  = nbase + wn * 16 + (lane & 15);
  const int mo = mbase + wm * 16 + ((lane < 16) ? 0 : 8);
  const float bn = bias[n];
#pragma unroll
  for (int r = 0; r < 8; ++r) {
    const int m = mo + r;
    float v = acc[r] + bn;
    if (RESID) v += resid[(size_t)m * N + n];
    if (ACT == 1) v = 0.5f * v * (1.f + erff(v * 0.70710678118f));  // exact GELU
    if (OUTF16) ((_Float16*)outp)[(size_t)m * N + n] = (_Float16)v;
    else        ((float*)outp)[(size_t)m * N + n] = v;
  }
}

// ---------------------------------------------------------------------------
// 4) Flash attention: wave owns 16 query rows; 32-key tiles; online softmax.
//    qkv f16 [S,384] (Q|K|V each 128 = 4 heads x 32). ctx f16 [S,128].
//    Fully wave-independent: per-wave LDS slices, no block barriers.
// ---------------------------------------------------------------------------
__global__ void attention_kernel(const _Float16* __restrict__ qkv,
                                 _Float16* __restrict__ ctx,
                                 const int* __restrict__ d_sele) {
  __shared__ _Float16 Vl[8][32][32];  // per-wave V tile, row-major [key][dh]
  __shared__ _Float16 Pl[8][16][32];  // per-wave probs [qrow][key]
  const int S    = SFIX;
  const int tid  = threadIdx.x;
  const int lane = tid & 31;
  const int w    = tid >> 5;
  const int h    = blockIdx.y;
  const int qbase = (blockIdx.x * 8 + w) * 16;
  const int sele  = *d_sele;
  const int nn   = lane & 15;
  const bool lo  = lane < 16;
  const int moff = lo ? 0 : 8;
  const int kb   = lo ? 0 : 8;
  const int koff = lo ? 0 : 16;
  const float scale = 0.17677669529663687f;  // 1/sqrt(32)

  // Q as A-fragment (held for whole loop)
  v16h aq;
  {
    const _Float16* qp = qkv + (size_t)(qbase + nn) * 384 + h * 32;
#pragma unroll
    for (int i = 0; i < 16; ++i) aq[i] = qp[koff + i];
  }
  v8f o0 = {}, o1 = {};
  float mrow[8], lrow[8];
#pragma unroll
  for (int r = 0; r < 8; ++r) { mrow[r] = -INFINITY; lrow[r] = 0.f; }

  for (int kt = 0; kt < S / 32; ++kt) {
    const int kbase = kt * 32;
    // Async-stage V tile: lane copies its key row (64B) as 4x16B chunks
    {
      const _Float16* vp = qkv + (size_t)(kbase + lane) * 384 + 256 + h * 32;
#pragma unroll
      for (int j = 0; j < 4; ++j) async_cp16(vp + j * 8, &Vl[w][lane][j * 8]);
    }
    // K B-fragments straight from global: lane -> one key row, contiguous dh
    v16h bk0, bk1;
    {
      const _Float16* kp0 =
          qkv + (size_t)(kbase + nn) * 384 + 128 + h * 32 + koff;
      const _Float16* kp1 = kp0 + (size_t)16 * 384;
#pragma unroll
      for (int i = 0; i < 16; ++i) { bk0[i] = kp0[i]; bk1[i] = kp1[i]; }
      if (kt + 1 < S / 32) __builtin_prefetch(kp0 + (size_t)32 * 384, 0, 0);
    }
    v8f z = {};
    v8f s0 = wmma_f16(aq, bk0, z);   // keys kbase .. kbase+15
    v8f s1 = wmma_f16(aq, bk1, z);   // keys kbase+16 .. kbase+31

    const int key0 = kbase + nn;
    const int key1 = kbase + 16 + nn;
#pragma unroll
    for (int r = 0; r < 8; ++r) {
      float v0 = (key0 < sele) ? s0[r] * scale : -INFINITY;
      float v1 = (key1 < sele) ? s1[r] * scale : -INFINITY;
      const float mx = dpp_max16(fmaxf(v0, v1));
      const float mn = fmaxf(mrow[r], mx);
      float p0 = 0.f, p1 = 0.f, corr = 1.f;
      if (mn != -INFINITY) {
        corr = expf(mrow[r] - mn);   // exp(-inf)=0 on first live tile
        p0   = expf(v0 - mn);
        p1   = expf(v1 - mn);
      }
      const float rs = dpp_sum16(p0 + p1);
      lrow[r] = lrow[r] * corr + rs;
      mrow[r] = mn;
      o0[r] *= corr;
      o1[r] *= corr;
      Pl[w][moff + r][nn]      = (_Float16)p0;   // C layout -> [row][key]
      Pl[w][moff + r][16 + nn] = (_Float16)p1;
    }
    // P back in A-fragment layout (same-wave DS ops are in-order)
    v16h pa;
#pragma unroll
    for (int i = 0; i < 16; ++i) pa[i] = Pl[w][nn][koff + i];
    // V B-fragments from async-staged tile: B[k=key][n=dh]
    wait_async0();
    v16h bv0, bv1;
#pragma unroll
    for (int i = 0; i < 16; ++i) {
      bv0[i] = Vl[w][koff + i][nn];
      bv1[i] = Vl[w][koff + i][16 + nn];
    }
    o0 = wmma_f16(pa, bv0, o0);
    o1 = wmma_f16(pa, bv1, o1);
  }
#pragma unroll
  for (int r = 0; r < 8; ++r) {
    const float inv = (lrow[r] > 0.f) ? 1.f / lrow[r] : 0.f;
    const int m = qbase + moff + r;
    ctx[(size_t)m * HID + h * 32 + nn]      = (_Float16)(o0[r] * inv);
    ctx[(size_t)m * HID + h * 32 + 16 + nn] = (_Float16)(o1[r] * inv);
  }
}

// ---------------------------------------------------------------------------
// 5) Decoders collapsed: colsum over unmasked rows, then 416 dot-128 products.
// ---------------------------------------------------------------------------
__global__ void zero_colsum_kernel(float* __restrict__ colsum) {
  colsum[threadIdx.x] = 0.f;
}

__global__ void colsum_kernel(const float* __restrict__ ef,
                              float* __restrict__ colsum,
                              const int* __restrict__ d_sele) {
  const int h  = threadIdx.x;          // 0..127
  const int r0 = blockIdx.x * 256;
  const int sele = *d_sele;
  float acc = 0.f;
  for (int r = 0; r < 256; ++r) {
    const int s = r0 + r;
    if (s < sele) acc += ef[(size_t)s * HID + h];
  }
  atomicAdd(&colsum[h], acc);
}

__global__ void decode_kernel(const float* __restrict__ colsum,
                              const float* __restrict__ Ws, const float* __restrict__ bs,
                              const float* __restrict__ Ww, const float* __restrict__ bw,
                              const float* __restrict__ Wc, const float* __restrict__ bc,
                              const int* __restrict__ d_len,
                              const int* __restrict__ d_sele,
                              float* __restrict__ coeS,
                              float* __restrict__ coeW,
                              float* __restrict__ coeC) {
  __shared__ float csh[HID];
  __shared__ float co[2 * N_COE * LAYERS + N_SCALES * LAYERS];  // 416
  const int t = threadIdx.x;
  if (t < HID) csh[t] = colsum[t];
  __syncthreads();
  const float len_f = (float)(*d_len) + 1e-8f;
  const float seln  = (float)(*d_sele);
  if (t < 416) {
    const float* Wp; float bb;
    if (t < 200)      { Wp = Ws + (size_t)t * HID;         bb = bs[t]; }
    else if (t < 400) { int j = t - 200; Wp = Ww + (size_t)j * HID; bb = bw[j]; }
    else              { int j = t - 400; Wp = Wc + (size_t)j * HID; bb = bc[j]; }
    float acc = 0.f;
#pragma unroll 4
    for (int k = 0; k < HID; ++k) acc = fmaf(csh[k], Wp[k], acc);
    co[t] = (acc + seln * bb) / len_f;
  }
  __syncthreads();
  if (t < 8) {  // softmax over c for (scaling|wavelet, layer)
    const int dec = t >> 2, l = t & 3;
    const float* src = co + dec * (N_COE * LAYERS);
    float* dst = (dec == 0) ? coeS : coeW;
    float mx = -INFINITY;
    for (int c = 0; c < N_COE; ++c) mx = fmaxf(mx, src[c * LAYERS + l]);
    float sum = 0.f;
    for (int c = 0; c < N_COE; ++c) sum += expf(src[c * LAYERS + l] - mx);
    const float inv = 1.f / sum;
    for (int c = 0; c < N_COE; ++c)
      dst[l * N_COE + c] = expf(src[c * LAYERS + l] - mx) * inv;
  } else if (t < 24) {  // sigmoid * scale_scale for coe_scales
    const int idx = t - 8;
    const int l = idx >> 2, c = idx & 3;
    const float ss = (c == 0) ? 0.5f : ((c == 1) ? 1.f : 10.f);
    const float v = co[2 * N_COE * LAYERS + c * LAYERS + l];
    coeC[l * N_SCALES + c] = ss / (1.f + expf(-v));
  }
}

// ---------------------------------------------------------------------------
// 6) Chebyshev bases (odd->scaling, even->wavelet), dot coeffs, L2 normalize.
// ---------------------------------------------------------------------------
__global__ void wavegen_final_kernel(const float* __restrict__ e,
                                     const float* __restrict__ coeS,
                                     const float* __restrict__ coeW,
                                     const float* __restrict__ coeC,
                                     float* __restrict__ out) {
  __shared__ float cs[N_COE], cw[N_COE], csc[N_SCALES];
  const int l = blockIdx.y;
  const int t = threadIdx.x;
  if (t < N_COE) { cs[t] = coeS[l * N_COE + t]; cw[t] = coeW[l * N_COE + t]; }
  if (t < N_SCALES) csc[t] = coeC[l * N_SCALES + t];
  __syncthreads();
  const int s = blockIdx.x * blockDim.x + t;
  const float ev = e[s];
  float v[1 + N_SCALES];
  { // scaling: odd Chebyshev orders of (e - 1)
    const float y = ev - 1.f;
    float te = 1.f, to = y;
    float acc = cs[0] * 0.5f * (1.f - to);
    for (int c = 1; c < N_COE; ++c) {
      te = 2.f * y * to - te;
      to = 2.f * y * te - to;
      acc += cs[c] * 0.5f * (1.f - to);
    }
    v[0] = acc;
  }
#pragma unroll
  for (int d = 0; d < N_SCALES; ++d) {  // wavelet: even orders of (e*scale - 1)
    const float fs = ev * csc[d];
    const float y = (fs > 2.f) ? -1.f : (fs - 1.f);
    float te = 1.f, to = y;
    float acc = 0.f;  // c=0 term: 0.5*(1 - T0) = 0
    for (int c = 1; c < N_COE; ++c) {
      te = 2.f * y * to - te;
      to = 2.f * y * te - to;
      acc += cw[c] * 0.5f * (1.f - te);
    }
    v[1 + d] = acc;
  }
  float nsq = 0.f;
#pragma unroll
  for (int i = 0; i < 5; ++i) nsq += v[i] * v[i];
  const float inv = 1.f / (sqrtf(nsq) + 1e-8f);
  const size_t base = ((size_t)l * SFIX + s) * 5;
#pragma unroll
  for (int i = 0; i < 5; ++i) out[base + i] = v[i] * inv;
}

// ---------------------------------------------------------------------------
extern "C" void kernel_launch(void* const* d_in, const int* in_sizes, int n_in,
                              void* d_out, int out_size, void* d_ws, size_t ws_size,
                              hipStream_t stream) {
  const float* eigval = (const float*)d_in[0];
  const float* sine_W = (const float*)d_in[1];
  const float* sine_b = (const float*)d_in[2];
  const float* ln1_g  = (const float*)d_in[3];
  const float* ln1_b  = (const float*)d_in[4];
  const float* ln2_g  = (const float*)d_in[5];
  const float* ln2_b  = (const float*)d_in[6];
  const float* W_in   = (const float*)d_in[7];
  const float* b_in   = (const float*)d_in[8];
  const float* W_out  = (const float*)d_in[9];
  const float* b_out  = (const float*)d_in[10];
  const float* ffn_W1 = (const float*)d_in[11];
  const float* ffn_b1 = (const float*)d_in[12];
  const float* ffn_W2 = (const float*)d_in[13];
  const float* ffn_b2 = (const float*)d_in[14];
  const float* dsW = (const float*)d_in[15];
  const float* dsb = (const float*)d_in[16];
  const float* dwW = (const float*)d_in[17];
  const float* dwb = (const float*)d_in[18];
  const float* dcW = (const float*)d_in[19];
  const float* dcb = (const float*)d_in[20];
  const int* d_len  = (const int*)d_in[21];
  const int* d_sele = (const int*)d_in[22];
  float* out = (float*)d_out;
  (void)in_sizes; (void)n_in; (void)out_size; (void)ws_size;

  const int S = SFIX;
  char* ws = (char*)d_ws;
  size_t off = 0;
  auto alloc = [&](size_t bytes) -> void* {
    void* p = (void*)(ws + off);
    off += (bytes + 255) & ~(size_t)255;
    return p;
  };
  float*    eig    = (float*)   alloc((size_t)S * HID * 4);
  _Float16* xln    = (_Float16*)alloc((size_t)S * HID * 2);
  _Float16* qkv    = (_Float16*)alloc((size_t)S * 3 * HID * 2);
  _Float16* ctx    = (_Float16*)alloc((size_t)S * HID * 2);
  float*    eig2   = (float*)   alloc((size_t)S * HID * 4);
  _Float16* yln    = (_Float16*)alloc((size_t)S * HID * 2);
  _Float16* h1     = (_Float16*)alloc((size_t)S * HID * 2);
  float*    eigf   = (float*)   alloc((size_t)S * HID * 4);
  float*    colsum = (float*)   alloc(HID * 4);
  float*    coeS   = (float*)   alloc(LAYERS * N_COE * 4);
  float*    coeW   = (float*)   alloc(LAYERS * N_COE * 4);
  float*    coeC   = (float*)   alloc(LAYERS * N_SCALES * 4);
  _Float16* Wh_in  = (_Float16*)alloc((size_t)3 * HID * HID * 2);
  _Float16* Wh_out = (_Float16*)alloc((size_t)HID * HID * 2);
  _Float16* Wh_f1  = (_Float16*)alloc((size_t)HID * HID * 2);
  _Float16* Wh_f2  = (_Float16*)alloc((size_t)HID * HID * 2);

  // Weight conversions (independent of activations; front-load them)
  f32_to_f16_kernel<<<(3 * HID * HID) / 256, 256, 0, stream>>>(W_in, Wh_in,
                                                               3 * HID * HID);
  f32_to_f16_kernel<<<(HID * HID) / 256, 256, 0, stream>>>(W_out, Wh_out,
                                                           HID * HID);
  f32_to_f16_kernel<<<(HID * HID) / 256, 256, 0, stream>>>(ffn_W1, Wh_f1,
                                                           HID * HID);
  f32_to_f16_kernel<<<(HID * HID) / 256, 256, 0, stream>>>(ffn_W2, Wh_f2,
                                                           HID * HID);

  sine_encode_kernel<<<S, HID, 0, stream>>>(eigval, sine_W, sine_b, eig);
  layernorm_kernel<<<S / 8, 256, 0, stream>>>(eig, ln1_g, ln1_b, xln);
  gemm_xWT_kernel<0, 0, 1><<<dim3(384 / 32, S / 64), 256, 0, stream>>>(
      xln, Wh_in, b_in, nullptr, qkv, 3 * HID);
  attention_kernel<<<dim3(S / 128, NHEADS), 256, 0, stream>>>(qkv, ctx, d_sele);
  gemm_xWT_kernel<0, 1, 0><<<dim3(HID / 32, S / 64), 256, 0, stream>>>(
      ctx, Wh_out, b_out, eig, eig2, HID);
  layernorm_kernel<<<S / 8, 256, 0, stream>>>(eig2, ln2_g, ln2_b, yln);
  gemm_xWT_kernel<1, 0, 1><<<dim3(HID / 32, S / 64), 256, 0, stream>>>(
      yln, Wh_f1, ffn_b1, nullptr, h1, HID);
  gemm_xWT_kernel<0, 1, 0><<<dim3(HID / 32, S / 64), 256, 0, stream>>>(
      h1, Wh_f2, ffn_b2, eig2, eigf, HID);
  zero_colsum_kernel<<<1, HID, 0, stream>>>(colsum);
  colsum_kernel<<<S / 256, HID, 0, stream>>>(eigf, colsum, d_sele);
  decode_kernel<<<1, 512, 0, stream>>>(colsum, dsW, dsb, dwW, dwb, dcW, dcb,
                                       d_len, d_sele, coeS, coeW, coeC);
  wavegen_final_kernel<<<dim3(S / 256, LAYERS), 256, 0, stream>>>(
      eigval, coeS, coeW, coeC, out);
}
